// GCN_dev_38938173505667
// MI455X (gfx1250) — matrix operand order repaired and checked by reading, not compiled, not run on hardware
//
#include <hip/hip_runtime.h>
#include <hip/hip_bf16.h>

// ---------------------------------------------------------------------------
// GCN (2-layer GraphConv) for MI455X / gfx1250.
//  - edge passes use native f32 atomics (L2-resident working set)
//  - layer-1 GEMM uses v_wmma_f32_16x16x32_bf16, one wave32 per 16-row block
//  - layer-2 (256->1) projection is fused into the GEMM epilogue, so the
//    layer-2 edge pass moves 1 scalar/edge instead of 256 floats/edge and
//    h1 never goes to memory.
// ---------------------------------------------------------------------------

typedef __attribute__((ext_vector_type(16))) __bf16 v16bf;
typedef __attribute__((ext_vector_type(8)))  float  v8f;

static __device__ __forceinline__ __bf16 f2bf(float f) {
  unsigned u = __builtin_bit_cast(unsigned, f);
  u += 0x7FFFu + ((u >> 16) & 1u);            // round-to-nearest-even
  unsigned short h = (unsigned short)(u >> 16);
  return __builtin_bit_cast(__bf16, h);
}

// ---- zero workspace (grid-stride float4) ----------------------------------
__global__ void zero_f4(float4* __restrict__ p, long n4) {
  long i = blockIdx.x * (long)blockDim.x + threadIdx.x;
  long stride = (long)gridDim.x * blockDim.x;
  float4 z = make_float4(0.f, 0.f, 0.f, 0.f);
  for (; i < n4; i += stride) p[i] = z;
}

// ---- degree accumulation ---------------------------------------------------
__global__ void degree_k(const int* __restrict__ src, const int* __restrict__ dst,
                         float* __restrict__ deg_out, float* __restrict__ deg_in, int E) {
  int e = blockIdx.x * blockDim.x + threadIdx.x;
  if (e < E) {
    unsafeAtomicAdd(&deg_out[src[e]], 1.0f);
    unsafeAtomicAdd(&deg_in[dst[e]], 1.0f);
  }
}

// ---- deg -> rsqrt(max(deg,1)) in place ------------------------------------
__global__ void norm_k(float* __restrict__ deg_out, float* __restrict__ deg_in, int N) {
  int i = blockIdx.x * blockDim.x + threadIdx.x;
  if (i < N) {
    deg_out[i] = rsqrtf(fmaxf(deg_out[i], 1.0f));
    deg_in[i]  = rsqrtf(fmaxf(deg_in[i], 1.0f));
  }
}

// ---- layer-1 edge scatter: agg1[dst] += x[src] * norm_out[src] -------------
// one wave per edge, each lane handles 4 features (float4 gather, 4 atomics)
__global__ void scatter1_k(const float* __restrict__ x,
                           const int* __restrict__ src, const int* __restrict__ dst,
                           const float* __restrict__ norm_out,
                           float* __restrict__ agg1, int E) {
  long tid = blockIdx.x * (long)blockDim.x + threadIdx.x;
  int e = (int)(tid >> 5);
  if (e >= E) return;
  int f = ((int)tid & 31) * 4;
  int s = src[e], d = dst[e];
  float no = norm_out[s];
  float4 xv = *(const float4*)(x + (size_t)s * 128 + f);
  float* b = agg1 + (size_t)d * 128 + f;
  unsafeAtomicAdd(b + 0, xv.x * no);
  unsafeAtomicAdd(b + 1, xv.y * no);
  unsafeAtomicAdd(b + 2, xv.z * no);
  unsafeAtomicAdd(b + 3, xv.w * no);
}

// ---- pre-swizzle W1 (f32 row-major [128,256]) into WMMA B fragments --------
// layout: frag[(t*4+kk)*32 + lane] = 16 contiguous bf16:
//   lanes 0-15 : N = t*16+lane,     K = kk*32 + {0..7, 16..23}
//   lanes 16-31: N = t*16+lane-16,  K = kk*32 + {8..15, 24..31}
__global__ void w1prep_k(const float* __restrict__ W1, __bf16* __restrict__ w1f) {
  int tid = blockIdx.x * blockDim.x + threadIdx.x;     // 0..2047
  if (tid >= 16 * 4 * 32) return;
  int lane = tid & 31;
  int kk = (tid >> 5) & 3;
  int t  = tid >> 7;
  int n  = t * 16 + (lane & 15);
  int kb = kk * 32 + ((lane >> 4) ? 8 : 0);
  __bf16* outp = w1f + (size_t)tid * 16;
#pragma unroll
  for (int e = 0; e < 16; ++e) outp[e] = f2bf(W1[(size_t)(kb + e + (e >= 8 ? 8 : 0)) * 256 + n]);
}

// ---- WMMA GEMM: h1 = relu((agg1*norm_in) @ W1 + b1); fused z = n_out*(h1.W2)
// one wave32 per 16-row block; A fragments held in registers across 16 N-tiles
__global__ void __launch_bounds__(32)
gemm_fused_k(const float* __restrict__ agg1,
             const float* __restrict__ norm_in, const float* __restrict__ norm_out,
             const __bf16* __restrict__ w1f,
             const float* __restrict__ b1, const float* __restrict__ W2,
             float* __restrict__ z, int N) {
  int lane = threadIdx.x;
  int rowBase = blockIdx.x * 16;
  int row = rowBase + (lane & 15);        // A-matrix: each lane owns one row M
  float nin = norm_in[row];

  // A fragments: 16x32 bf16 per K-step, per ISA layout:
  //   lanes 0-15 hold K = {kb..kb+7, kb+16..kb+23} with kb = kk*32
  //   lanes 16-31 same with kb = kk*32+8
  v16bf a[4];
  int kb0 = (lane >> 4) ? 8 : 0;
#pragma unroll
  for (int kk = 0; kk < 4; ++kk) {
    int kb = kk * 32 + kb0;
    const float4* ap = (const float4*)(agg1 + (size_t)row * 128 + kb);
    float4 c0 = ap[0], c1 = ap[1], c2 = ap[4], c3 = ap[5];
    float tmp[16] = {c0.x, c0.y, c0.z, c0.w, c1.x, c1.y, c1.z, c1.w,
                     c2.x, c2.y, c2.z, c2.w, c3.x, c3.y, c3.z, c3.w};
#pragma unroll
    for (int i = 0; i < 16; ++i) a[kk][i] = f2bf(tmp[i] * nin);
  }

  const v16bf* bp = (const v16bf*)w1f;
  float zacc[8];
#pragma unroll
  for (int r = 0; r < 8; ++r) zacc[r] = 0.f;

  for (int t = 0; t < 16; ++t) {          // 16 N-tiles of 16 cols each
    v8f c = {};
#pragma unroll
    for (int kk = 0; kk < 4; ++kk) {
      v16bf b = bp[(t * 4 + kk) * 32 + lane];
      c = __builtin_amdgcn_wmma_f32_16x16x32_bf16(
          /*neg_a=*/false, a[kk], /*neg_b=*/false, b,
          /*c_mod=*/(short)0, c, /*reuse_a=*/false, /*reuse_b=*/false);
    }
    // C/D layout: VGPR r, lanes 0-15 -> (M=r, N=lane); lanes 16-31 -> (M=r+8)
    int n = t * 16 + (lane & 15);
    float bias = b1[n];
    float w2v = W2[n];
#pragma unroll
    for (int r = 0; r < 8; ++r) {
      float h = fmaxf(c[r] + bias, 0.f);  // relu(h1)
      zacc[r] += h * w2v;                 // fused 256->1 projection
    }
  }

  // reduce over the 16 lanes of each half (wave32: xor 1,2,4,8 stays in half)
#pragma unroll
  for (int r = 0; r < 8; ++r) {
    float v = zacc[r];
    v += __shfl_xor(v, 1);
    v += __shfl_xor(v, 2);
    v += __shfl_xor(v, 4);
    v += __shfl_xor(v, 8);
    zacc[r] = v;
  }
  if ((lane & 15) == 0) {
    int rb = rowBase + ((lane >> 4) ? 8 : 0);
#pragma unroll
    for (int r = 0; r < 8; ++r) {
      int rr = rb + r;
      if (rr < N) z[rr] = zacc[r] * norm_out[rr];   // pre-scale by D_out^-1/2
    }
  }
}

// ---- layer-2 edge scatter (scalar per edge) --------------------------------
__global__ void scatter2_k(const int* __restrict__ src, const int* __restrict__ dst,
                           const float* __restrict__ z, float* __restrict__ zagg, int E) {
  int e = blockIdx.x * blockDim.x + threadIdx.x;
  if (e < E) unsafeAtomicAdd(&zagg[dst[e]], z[src[e]]);
}

// ---- output: sigmoid(norm_in * zagg + b2) ----------------------------------
__global__ void out_k(const float* __restrict__ zagg, const float* __restrict__ norm_in,
                      const float* __restrict__ b2, float* __restrict__ out, int N) {
  int i = blockIdx.x * blockDim.x + threadIdx.x;
  if (i < N) {
    float v = norm_in[i] * zagg[i] + b2[0];
    out[i] = 1.0f / (1.0f + __expf(-v));
  }
}

extern "C" void kernel_launch(void* const* d_in, const int* in_sizes, int n_in,
                              void* d_out, int out_size, void* d_ws, size_t ws_size,
                              hipStream_t stream) {
  const float* x  = (const float*)d_in[0];   // [N,128]
  const int*   src = (const int*)d_in[1];    // [E]
  const int*   dst = (const int*)d_in[2];    // [E]
  const float* W1 = (const float*)d_in[3];   // [128,256]
  const float* b1 = (const float*)d_in[4];   // [256]
  const float* W2 = (const float*)d_in[5];   // [256,1]
  const float* b2 = (const float*)d_in[6];   // [1]
  const int N = in_sizes[0] / 128;
  const int E = in_sizes[1];
  float* out = (float*)d_out;

  // workspace layout (floats): deg_out[N] deg_in[N] zagg[N] agg1[N*128] z[N] | w1frag(bf16 64KB)
  float* ws      = (float*)d_ws;
  float* deg_out = ws;                       // becomes norm_out after norm_k
  float* deg_in  = deg_out + N;              // becomes norm_in
  float* zagg    = deg_in + N;
  float* agg1    = zagg + N;
  float* zbuf    = agg1 + (size_t)N * 128;
  __bf16* w1f    = (__bf16*)(zbuf + N);      // 32-byte aligned for N=100000

  // zero everything accumulated into (131*N floats)
  long zeroN = (long)N * 131;
  zero_f4<<<4096, 256, 0, stream>>>((float4*)ws, (zeroN + 3) / 4);

  degree_k<<<(E + 255) / 256, 256, 0, stream>>>(src, dst, deg_out, deg_in, E);
  norm_k<<<(N + 255) / 256, 256, 0, stream>>>(deg_out, deg_in, N);

  long st1 = (long)E * 32;                   // one lane per 4 features per edge
  scatter1_k<<<(unsigned)((st1 + 255) / 256), 256, 0, stream>>>(
      x, src, dst, deg_out, agg1, E);

  w1prep_k<<<8, 256, 0, stream>>>(W1, w1f);

  gemm_fused_k<<<(N + 15) / 16, 32, 0, stream>>>(
      agg1, deg_in, deg_out, w1f, b1, W2, zbuf, N);

  scatter2_k<<<(E + 255) / 256, 256, 0, stream>>>(src, dst, zbuf, zagg, E);
  out_k<<<(N + 255) / 256, 256, 0, stream>>>(zagg, deg_in, b2, out, N);
}
